// DeepDiveLSTM_86543591014817
// MI455X (gfx1250) — matrix operand order repaired and checked by reading, not compiled, not run on hardware
//
#include <hip/hip_runtime.h>
#include <hip/hip_bf16.h>
#include <cstdint>

// Problem dims (fixed by the reference)
#define S_SEQ 512
#define BATCH 256
#define IN_DIM 256
#define HID 512
#define OUT_DIM 256
#define GATES (4 * HID)   // 2048

typedef __attribute__((ext_vector_type(16))) __bf16 v16bf;
typedef __attribute__((ext_vector_type(8)))  float  v8f;

union Frag16 {
    v16bf v;
    uint32_t u[8];
};

__device__ __forceinline__ float sigmoidf(float x) {
    return 1.0f / (1.0f + __expf(-x));
}

// A-matrix fragment (16x32 bf16, M x K), row-major source with stride lda (elements).
// ISA layout: lane (m = lane&15, hf = lane>>4); VGPR v<4: K = hf*8 + 2v,2v+1 ;
//             VGPR v>=4: K = 16 + hf*8 + 2(v-4), +1.
__device__ __forceinline__ v16bf load_a16(const __hip_bfloat16* __restrict__ A,
                                          int lda, int row0, int k0, int m, int hf) {
    const __hip_bfloat16* base = A + (size_t)(row0 + m) * lda + k0 + hf * 8;
    Frag16 f;
    uint4 c0 = *reinterpret_cast<const uint4*>(base);        // K = hf*8 .. hf*8+7
    uint4 c1 = *reinterpret_cast<const uint4*>(base + 16);   // K = 16+hf*8 .. +7
    f.u[0] = c0.x; f.u[1] = c0.y; f.u[2] = c0.z; f.u[3] = c0.w;
    f.u[4] = c1.x; f.u[5] = c1.y; f.u[6] = c1.z; f.u[7] = c1.w;
    return f.v;
}

// B-matrix fragment (32x16 bf16, K x N) sourced from row-major W[g][k] (stride ldb),
// so B[k][n] = W[col0+n][k0+k].
// ISA layout: lane (n = lane&15, hf = lane>>4); VGPR v<4: K = hf*8 + 2v,2v+1 ;
//             VGPR v>=4: K = 16 + hf*8 + 2(v-4), +1.
__device__ __forceinline__ v16bf load_b16(const __hip_bfloat16* __restrict__ W,
                                          int ldb, int col0, int k0, int n, int hf) {
    const __hip_bfloat16* base = W + (size_t)(col0 + n) * ldb + k0 + hf * 8;
    Frag16 f;
    uint4 c0 = *reinterpret_cast<const uint4*>(base);
    uint4 c1 = *reinterpret_cast<const uint4*>(base + 16);
    f.u[0] = c0.x; f.u[1] = c0.y; f.u[2] = c0.z; f.u[3] = c0.w;
    f.u[4] = c1.x; f.u[5] = c1.y; f.u[6] = c1.z; f.u[7] = c1.w;
    return f.v;
}

// Fused per-timestep kernel.
// Blocks [0,64): compute h_t, c_t from h_{t-1}  (16x32 = 512 tiles of 16x16, 8 waves/block)
// Blocks [64,96): compute head output for h_{t-1} (16x16 = 256 tiles of 16x16)
__global__ __launch_bounds__(256) void lstm_step_kernel(
    const __hip_bfloat16* __restrict__ x_t,     // [B, I] bf16
    const __hip_bfloat16* __restrict__ h_prev,  // [B, H] bf16
    __hip_bfloat16* __restrict__ h_next,        // [B, H] bf16 (ping-pong)
    float* __restrict__ c_state,                // [B, H] f32, in/out
    float* __restrict__ h_state32,              // [B, H] f32, out
    const __hip_bfloat16* __restrict__ Wih,     // [G, I] bf16
    const __hip_bfloat16* __restrict__ Whh,     // [G, H] bf16
    const float* __restrict__ bias,             // [G]  (b_ih + b_hh)
    const __hip_bfloat16* __restrict__ Wfc,     // [O, H] bf16
    const float* __restrict__ bfc,              // [O]
    const __hip_bfloat16* __restrict__ head_h,  // [B, H] bf16 (== h_{t-1})
    float* __restrict__ head_out,               // [B, O] f32 slice of d_out, or null
    int do_gates)
{
    const int wave = threadIdx.x >> 5;
    const int lane = threadIdx.x & 31;
    const int m    = lane & 15;     // M/N index within tile
    const int hf   = lane >> 4;     // lane-half

    if (blockIdx.x < 64) {
        // ------------------ gate GEMM + cell update ------------------
        if (!do_gates) return;
        const int tile = blockIdx.x * 8 + wave;     // 0..511
        const int row0 = (tile & 15) * 16;          // batch rows
        const int col0 = (tile >> 4) * 16;          // hidden cols

        v8f acc[4];
#pragma unroll
        for (int g = 0; g < 4; ++g)
#pragma unroll
            for (int e = 0; e < 8; ++e) acc[g][e] = 0.0f;

        // input projection: K over I
        for (int k0 = 0; k0 < IN_DIM; k0 += 32) {
            v16bf a = load_a16(x_t, IN_DIM, row0, k0, m, hf);
#pragma unroll
            for (int g = 0; g < 4; ++g) {
                v16bf b = load_b16(Wih, IN_DIM, g * HID + col0, k0, m, hf);
                acc[g] = __builtin_amdgcn_wmma_f32_16x16x32_bf16(
                    false, a, false, b, (short)0, acc[g], false, false);
            }
        }
        // recurrent projection: K over H
        for (int k0 = 0; k0 < HID; k0 += 32) {
            v16bf a = load_a16(h_prev, HID, row0, k0, m, hf);
#pragma unroll
            for (int g = 0; g < 4; ++g) {
                v16bf b = load_b16(Whh, HID, g * HID + col0, k0, m, hf);
                acc[g] = __builtin_amdgcn_wmma_f32_16x16x32_bf16(
                    false, a, false, b, (short)0, acc[g], false, false);
            }
        }

        const int j = col0 + m;   // hidden column for this lane
        const float bi = bias[j];
        const float bf = bias[HID + j];
        const float bg = bias[2 * HID + j];
        const float bo = bias[3 * HID + j];

#pragma unroll
        for (int v = 0; v < 8; ++v) {
            const int brow = row0 + v + 8 * hf;      // batch row per C/D layout
            const size_t idx = (size_t)brow * HID + j;
            const float ig = sigmoidf(acc[0][v] + bi);
            const float fg = sigmoidf(acc[1][v] + bf);
            const float gg = tanhf(acc[2][v] + bg);
            const float og = sigmoidf(acc[3][v] + bo);
            const float cc = fg * c_state[idx] + ig * gg;
            c_state[idx] = cc;
            const float hh = og * tanhf(cc);
            h_state32[idx] = hh;
            h_next[idx] = __float2bfloat16(hh);
        }
    } else {
        // ------------------ head GEMM for h_{t-1} ------------------
        if (head_out == nullptr) return;
        const int tile = (blockIdx.x - 64) * 8 + wave;  // 0..255
        const int row0 = (tile & 15) * 16;              // batch rows
        const int col0 = (tile >> 4) * 16;              // output cols

        v8f acc;
#pragma unroll
        for (int e = 0; e < 8; ++e) acc[e] = 0.0f;

        for (int k0 = 0; k0 < HID; k0 += 32) {
            v16bf a = load_a16(head_h, HID, row0, k0, m, hf);
            v16bf b = load_b16(Wfc, HID, col0, k0, m, hf);
            acc = __builtin_amdgcn_wmma_f32_16x16x32_bf16(
                false, a, false, b, (short)0, acc, false, false);
        }

        const float bo = bfc[col0 + m];
#pragma unroll
        for (int v = 0; v < 8; ++v) {
            const int brow = row0 + v + 8 * hf;
            head_out[(size_t)brow * OUT_DIM + col0 + m] = acc[v] + bo;
        }
    }
}

// fp32 -> bf16 elementwise conversion (grid-stride)
__global__ void cvt_f32_to_bf16(const float* __restrict__ src,
                                __hip_bfloat16* __restrict__ dst, int n) {
    int i = blockIdx.x * blockDim.x + threadIdx.x;
    const int stride = gridDim.x * blockDim.x;
    for (; i < n; i += stride) dst[i] = __float2bfloat16(src[i]);
}

// bias = b_ih + b_hh
__global__ void sum_bias_kernel(const float* __restrict__ a,
                                const float* __restrict__ b,
                                float* __restrict__ dst, int n) {
    int i = blockIdx.x * blockDim.x + threadIdx.x;
    if (i < n) dst[i] = a[i] + b[i];
}

extern "C" void kernel_launch(void* const* d_in, const int* in_sizes, int n_in,
                              void* d_out, int out_size, void* d_ws, size_t ws_size,
                              hipStream_t stream) {
    (void)in_sizes; (void)n_in; (void)out_size; (void)ws_size;

    const float* x_seq  = (const float*)d_in[0];  // [S, B, I]
    const float* Wih32  = (const float*)d_in[1];  // [4H, I]
    const float* Whh32  = (const float*)d_in[2];  // [4H, H]
    const float* bih    = (const float*)d_in[3];  // [4H]
    const float* bhh    = (const float*)d_in[4];  // [4H]
    const float* Wfc32  = (const float*)d_in[5];  // [O, H]
    const float* bfc    = (const float*)d_in[6];  // [O]
    float* out = (float*)d_out;                   // [S,B,O] ++ h_f[B,H] ++ c_f[B,H]

    // ---- workspace layout (all chunks 256B aligned by construction) ----
    char* ws = (char*)d_ws;
    size_t off = 0;
    __hip_bfloat16* x_bf  = (__hip_bfloat16*)(ws + off); off += (size_t)S_SEQ * BATCH * IN_DIM * 2;  // 64 MiB
    __hip_bfloat16* WihB  = (__hip_bfloat16*)(ws + off); off += (size_t)GATES * IN_DIM * 2;          // 1 MiB
    __hip_bfloat16* WhhB  = (__hip_bfloat16*)(ws + off); off += (size_t)GATES * HID * 2;             // 2 MiB
    __hip_bfloat16* WfcB  = (__hip_bfloat16*)(ws + off); off += (size_t)OUT_DIM * HID * 2;           // 256 KiB
    float*          bias  = (float*)(ws + off);          off += (size_t)GATES * 4;                   // 8 KiB
    __hip_bfloat16* hb0   = (__hip_bfloat16*)(ws + off); off += (size_t)BATCH * HID * 2;             // 256 KiB
    __hip_bfloat16* hb1   = (__hip_bfloat16*)(ws + off); off += (size_t)BATCH * HID * 2;             // 256 KiB
    float*          c32   = (float*)(ws + off);          off += (size_t)BATCH * HID * 4;             // 512 KiB
    float*          h32   = (float*)(ws + off);          off += (size_t)BATCH * HID * 4;             // 512 KiB

    // ---- one-time (per launch) prep: converts + zero init ----
    cvt_f32_to_bf16<<<2048, 256, 0, stream>>>(x_seq, x_bf, S_SEQ * BATCH * IN_DIM);
    cvt_f32_to_bf16<<<512, 256, 0, stream>>>(Wih32, WihB, GATES * IN_DIM);
    cvt_f32_to_bf16<<<512, 256, 0, stream>>>(Whh32, WhhB, GATES * HID);
    cvt_f32_to_bf16<<<256, 256, 0, stream>>>(Wfc32, WfcB, OUT_DIM * HID);
    sum_bias_kernel<<<(GATES + 255) / 256, 256, 0, stream>>>(bih, bhh, bias, GATES);
    hipMemsetAsync(hb0, 0, (size_t)BATCH * HID * 2, stream);   // h_{-1} = 0
    hipMemsetAsync(c32, 0, (size_t)BATCH * HID * 4, stream);   // c_{-1} = 0

    // ---- sequential scan: one fused kernel per timestep ----
    // Step t: blocks[0,64) compute h_t from h_{t-1}; blocks[64,96) compute head(h_{t-1}).
    for (int t = 0; t < S_SEQ; ++t) {
        const __hip_bfloat16* hprev = (t & 1) ? hb1 : hb0;
        __hip_bfloat16*       hnext = (t & 1) ? hb0 : hb1;
        float* head_out = (t > 0) ? (out + (size_t)(t - 1) * BATCH * OUT_DIM) : nullptr;
        lstm_step_kernel<<<96, 256, 0, stream>>>(
            x_bf + (size_t)t * BATCH * IN_DIM, hprev, hnext, c32, h32,
            WihB, WhhB, bias, WfcB, bfc,
            /*head_h=*/hprev, head_out, /*do_gates=*/1);
    }
    // Final head for h_{S-1} (lives in hb0: t=511 odd wrote hnext=hb0).
    lstm_step_kernel<<<96, 256, 0, stream>>>(
        x_bf, hb0, hb1, c32, h32, WihB, WhhB, bias, WfcB, bfc,
        /*head_h=*/hb0, out + (size_t)(S_SEQ - 1) * BATCH * OUT_DIM, /*do_gates=*/0);

    // Final states into d_out tail: [outputs | h_f | c_f]
    float* tail = out + (size_t)S_SEQ * BATCH * OUT_DIM;
    hipMemcpyAsync(tail, h32, (size_t)BATCH * HID * 4, hipMemcpyDeviceToDevice, stream);
    hipMemcpyAsync(tail + (size_t)BATCH * HID, c32, (size_t)BATCH * HID * 4,
                   hipMemcpyDeviceToDevice, stream);
}
